// GroupContextModel2_32203664785696
// MI455X (gfx1250) — matrix operand order repaired and checked by reading, not compiled, not run on hardware
//
#include <hip/hip_runtime.h>
#include <hip/hip_bf16.h>
#include <math.h>
#include <stdint.h>

typedef __attribute__((ext_vector_type(16))) __bf16 v16bf;
typedef __attribute__((ext_vector_type(8)))  __bf16 v8bf;
typedef __attribute__((ext_vector_type(8)))  float  v8f;

union FragAB { v16bf v; v8bf h[2]; };

#define HW   128
#define NPOS 65536   // 4 * 128 * 128

// CDNA5 async copy: global -> LDS, no VGPR round trip, tracked by ASYNCcnt.
__device__ __forceinline__ void async_ld_b128(unsigned lds_off, const void* gptr) {
    asm volatile("global_load_async_to_lds_b128 %0, %1, off"
                 :: "v"(lds_off), "v"((unsigned long long)(uintptr_t)gptr)
                 : "memory");
}
__device__ __forceinline__ void wait_async0() {
    asm volatile("s_wait_asynccnt 0x0" ::: "memory");
}
__device__ __forceinline__ unsigned lds_off_of(const void* p) {
    return (unsigned)(uintptr_t)p;   // low 32 bits of generic addr = LDS offset
}

// ---------------- quantize + space_to_depth of q -> NHWC bf16 ----------------
__global__ void k_quant(const float* __restrict__ in, float* __restrict__ qout,
                        __bf16* __restrict__ qsd) {
    size_t i = (size_t)blockIdx.x * blockDim.x + threadIdx.x;
    if (i >= (size_t)4 * 128 * 256 * 256) return;
    float r = rintf(in[i]);                 // round half-to-even, matches jnp.round
    qout[i] = r;
    int X = i & 255;
    int Y = (int)((i >> 8) & 255);
    int c = (int)((i >> 16) & 127);
    int b = (int)(i >> 23);
    int oc = (c << 2) | ((Y & 1) << 1) | (X & 1);      // channel in [0,512)
    int y = Y >> 1, x = X >> 1;
    qsd[((((size_t)b * HW + y) * HW + x) << 9) + oc] = (__bf16)r;
}

// ------------- space_to_depth(condition) -> h (bf16 NHWC) and c (f32 NHWC) ----
__global__ void k_cond(const float* __restrict__ cond, __bf16* __restrict__ hbuf,
                       float* __restrict__ cbuf) {
    size_t i = (size_t)blockIdx.x * blockDim.x + threadIdx.x;
    if (i >= (size_t)4 * 32 * 256 * 256) return;
    float v = cond[i];
    int X = i & 255;
    int Y = (int)((i >> 8) & 255);
    int c = (int)((i >> 16) & 31);
    int b = (int)(i >> 21);
    int oc = (c << 2) | ((Y & 1) << 1) | (X & 1);      // [0,128)
    int y = Y >> 1, x = X >> 1;
    size_t p = ((size_t)b * HW + y) * HW + x;
    hbuf[(p << 7) + oc] = (__bf16)v;
    cbuf[(p << 7) + oc] = v;
}

// ----- reorder OIHW f32 weights -> [M][(r*3+s)*Cin + ci] bf16 (GEMM A matrix) -
__global__ void k_wconv(const float* __restrict__ src, __bf16* __restrict__ dst,
                        int Cin, int ksq, size_t total) {
    size_t i = (size_t)blockIdx.x * blockDim.x + threadIdx.x;
    if (i >= total) return;
    int K  = Cin * ksq;
    size_t m  = i / K;
    int    kd = (int)(i % K);
    int rs = kd / Cin, ci = kd % Cin;
    dst[i] = (__bf16)src[(m * Cin + ci) * (size_t)ksq + rs];
}

// ------------- pack a 128-channel tensor slice into the 256-ch concat buffer --
__global__ void k_pack(__bf16* __restrict__ cat, int dstOff,
                       const __bf16* __restrict__ src, int srcPitch, int srcOff,
                       int zero) {
    int i = blockIdx.x * blockDim.x + threadIdx.x;       // NPOS * 16 chunks of 8
    if (i >= NPOS * 16) return;
    int p = i >> 4, v = (i & 15) << 3;
    uint4 val = {0u, 0u, 0u, 0u};
    if (!zero) val = *(const uint4*)(src + (size_t)p * srcPitch + srcOff + v);
    *(uint4*)(cat + ((size_t)p << 8) + dstOff + v) = val;
}

// ---------------------------- WMMA implicit-GEMM conv ------------------------
// A = weights [M][K] bf16, B = activations NHWC [NPOS][256] bf16 (implicit 3x3
// halo when K3), C = f32. Double-buffered LDS tiles, staged with
// global_load_async_to_lds_b128 (ASYNCcnt). MODE 0: f32 (+bias) to outF[pos][M].
// MODE 1: leaky-relu -> bf16 to outH[pos][256].
template<int MODE, bool K3>
__global__ __launch_bounds__(256) void k_gemm(
    const __bf16* __restrict__ Wbf, const __bf16* __restrict__ act,
    const float* __restrict__ bias, float* __restrict__ outF,
    __bf16* __restrict__ outH, int M, int K)
{
    __shared__ __align__(16) __bf16 As[2][128][40];   // [buf][m][k]
    __shared__ __align__(16) __bf16 Bs[2][128][40];   // [buf][n][k]

    int tid  = threadIdx.x;
    int lane = tid & 31, ln = lane & 15, l16 = lane >> 4;
    int wv   = tid >> 5;
    int wm   = wv & 3;          // 4 waves tile M (32 rows each)
    int wn   = wv >> 2;         // 2 waves tile N (64 cols each)
    int n0   = blockIdx.x << 7;
    int m0   = blockIdx.y << 7;

    v8f acc[2][4] = {};

    int arow = tid >> 1, acb = (tid & 1) << 4;      // A stage: row, k sub-block
    int bn   = tid >> 1, bcb = (tid & 1) << 4;      // B stage: n, ch sub-block
    int p  = n0 + bn;
    int bb = p >> 14, py = (p >> 7) & 127, px = p & 127;

    auto stageA = [&](int k0, int buf) {
        const __bf16* g = Wbf + (size_t)(m0 + arow) * K + k0 + acb;
        async_ld_b128(lds_off_of(&As[buf][arow][acb]),     g);
        async_ld_b128(lds_off_of(&As[buf][arow][acb + 8]), g + 8);
    };
    auto stageB = [&](int k0, int buf) {
        int rs = k0 >> 8;                           // which 3x3 tap (0 for 1x1)
        int dy = 0, dx = 0;
        if (K3) { dy = rs / 3 - 1; dx = rs % 3 - 1; }
        int ci0 = k0 & 255;
        int sy = py + dy, sx = px + dx;
        if ((unsigned)sy < 128u && (unsigned)sx < 128u) {
            const __bf16* g = act +
                (((size_t)(bb * 128 + sy) * 128 + sx) << 8) + ci0 + bcb;
            async_ld_b128(lds_off_of(&Bs[buf][bn][bcb]),     g);
            async_ld_b128(lds_off_of(&Bs[buf][bn][bcb + 8]), g + 8);
        } else {
            uint4 z = {0u, 0u, 0u, 0u};
            *(uint4*)&Bs[buf][bn][bcb]     = z;     // halo: zero-fill via DS
            *(uint4*)&Bs[buf][bn][bcb + 8] = z;
        }
    };

    int nk = K >> 5;
    stageA(0, 0);
    stageB(0, 0);

    for (int it = 0; it < nk; ++it) {
        int cur = it & 1;
        wait_async0();            // this wave's tile landed in LDS
        __syncthreads();          // everyone's tile landed; prev buffer free
        if (it + 1 < nk) {        // prefetch next tile into the other buffer
            stageA((it + 1) << 5, cur ^ 1);
            stageB((it + 1) << 5, cur ^ 1);
        }

        // ---- fragments per CDNA5 16-bit WMMA lane layouts ----
        FragAB a[2], b[4];
        #pragma unroll
        for (int mi = 0; mi < 2; ++mi) {
            int m = (wm << 5) + (mi << 4) + ln;
            a[mi].h[0] = *(const v8bf*)&As[cur][m][(l16 << 3)];       // K=8*l16..
            a[mi].h[1] = *(const v8bf*)&As[cur][m][16 + (l16 << 3)];  // K=16+8*l16..
        }
        #pragma unroll
        for (int ni = 0; ni < 4; ++ni) {
            int n = (wn << 6) + (ni << 4) + ln;
            b[ni].h[0] = *(const v8bf*)&Bs[cur][n][(l16 << 3)];       // K=8*l16..
            b[ni].h[1] = *(const v8bf*)&Bs[cur][n][16 + (l16 << 3)];
        }
        #pragma unroll
        for (int mi = 0; mi < 2; ++mi)
            #pragma unroll
            for (int ni = 0; ni < 4; ++ni)
                acc[mi][ni] = __builtin_amdgcn_wmma_f32_16x16x32_bf16(
                    false, a[mi].v, false, b[ni].v,
                    (short)0, acc[mi][ni], false, false);
        // no trailing barrier: reuse of this buffer is gated by the barrier at
        // the top of the next iteration (all waves done reading by then)
    }

    // ---- epilogue: C/D layout lane n = ln, VGPR r -> m = r + 8*l16 ----
    #pragma unroll
    for (int mi = 0; mi < 2; ++mi) {
        int mg = m0 + (wm << 5) + (mi << 4) + (l16 << 3);
        #pragma unroll
        for (int ni = 0; ni < 4; ++ni) {
            int ng = n0 + (wn << 6) + (ni << 4) + ln;
            #pragma unroll
            for (int r = 0; r < 8; ++r) {
                float v = acc[mi][ni][r] + bias[mg + r];
                if (MODE == 0) {
                    outF[(size_t)ng * M + mg + r] = v;
                } else {
                    float lr = v >= 0.f ? v : 0.01f * v;
                    outH[((size_t)ng << 8) + mg + r] = (__bf16)lr;
                }
            }
        }
    }
}

// ------------------------------ ConvLSTM cell update -------------------------
__global__ void k_lstm(const float* __restrict__ gates, float* __restrict__ cbuf,
                       __bf16* __restrict__ hbuf) {
    size_t i = (size_t)blockIdx.x * blockDim.x + threadIdx.x;
    if (i >= (size_t)NPOS * 128) return;
    size_t p = i >> 7; int c = (int)(i & 127);
    const float* gp = gates + (p << 9);
    float ig = gp[c], fg = gp[128 + c], og = gp[256 + c], gg = gp[384 + c];
    float si = 1.f / (1.f + expf(-ig));
    float sf = 1.f / (1.f + expf(-fg));
    float so = 1.f / (1.f + expf(-og));
    float cn = sf * cbuf[i] + si * tanhf(gg);
    cbuf[i] = cn;
    hbuf[i] = (__bf16)(so * tanhf(cn));
}

// ------------------------- Gaussian CDF likelihood (NCHW out) ----------------
__global__ void k_lik(const float* __restrict__ cnd, const __bf16* __restrict__ qsd,
                      float* __restrict__ out, int g) {
    size_t i = (size_t)blockIdx.x * blockDim.x + threadIdx.x;
    if (i >= (size_t)NPOS * 128) return;
    size_t p = i >> 7; int c = (int)(i & 127);
    int b = (int)(p >> 14), y = (int)((p >> 7) & 127), x = (int)(p & 127);
    float mean  = cnd[(p << 8) + c];
    float scale = fmaxf(cnd[(p << 8) + 128 + c], 0.11f);
    float z = (float)qsd[(p << 9) + (g << 7) + c];
    float inv = 0.70710678118654752440f / scale;     // 1/(scale*sqrt(2))
    float up = 0.5f * (1.f + erff((z - mean + 0.5f) * inv));
    float lo = 0.5f * (1.f + erff((z - mean - 0.5f) * inv));
    out[(((size_t)b * 512 + (g << 7) + c) << 14) + (y << 7) + x] =
        fmaxf(up - lo, 1e-9f);
}

extern "C" void kernel_launch(void* const* d_in, const int* in_sizes, int n_in,
                              void* d_out, int out_size, void* d_ws, size_t ws_size,
                              hipStream_t stream) {
    const float* input     = (const float*)d_in[0];
    const float* condition = (const float*)d_in[1];
    const float* lstm_w    = (const float*)d_in[2];
    const float* lstm_b    = (const float*)d_in[3];
    const float* w1        = (const float*)d_in[4];
    const float* b1        = (const float*)d_in[5];
    const float* w2        = (const float*)d_in[6];
    const float* b2        = (const float*)d_in[7];
    const float* w3        = (const float*)d_in[8];
    const float* b3        = (const float*)d_in[9];

    float* qout = (float*)d_out;                 // [4,128,256,256]
    float* lik  = qout + (size_t)33554432;       // [4,512,128,128]

    char* ws = (char*)d_ws;
    size_t off = 0;
    auto alloc = [&](size_t bytes) {
        char* p = ws + off;
        off += (bytes + 255) & ~(size_t)255;
        return p;
    };
    __bf16* qsd   = (__bf16*)alloc((size_t)NPOS * 512 * 2);   // q, NHWC
    __bf16* hbuf  = (__bf16*)alloc((size_t)NPOS * 128 * 2);   // h state
    float*  cbuf  = (float*) alloc((size_t)NPOS * 128 * 4);   // c state
    __bf16* cat   = (__bf16*)alloc((size_t)NPOS * 256 * 2);   // concat input
    float*  gates = (float*) alloc((size_t)NPOS * 512 * 4);   // gates / cnd
    __bf16* t1    = (__bf16*)alloc((size_t)NPOS * 256 * 2);
    __bf16* t2    = (__bf16*)alloc((size_t)NPOS * 256 * 2);
    __bf16* Wl    = (__bf16*)alloc((size_t)512 * 2304 * 2);
    __bf16* W1    = (__bf16*)alloc((size_t)256 * 256 * 2);
    __bf16* W2    = (__bf16*)alloc((size_t)256 * 2304 * 2);
    __bf16* W3    = (__bf16*)alloc((size_t)256 * 256 * 2);
    float*  cnd   = gates;                        // reuse (no live overlap)

    dim3 blk(256);
    k_quant<<<(33554432 + 255) / 256, blk, 0, stream>>>(input, qout, qsd);
    k_cond <<<(8388608  + 255) / 256, blk, 0, stream>>>(condition, hbuf, cbuf);
    k_wconv<<<((512 * 2304) + 255) / 256, blk, 0, stream>>>(lstm_w, Wl, 256, 9, (size_t)512 * 2304);
    k_wconv<<<((256 * 256 ) + 255) / 256, blk, 0, stream>>>(w1, W1, 256, 1, (size_t)256 * 256);
    k_wconv<<<((256 * 2304) + 255) / 256, blk, 0, stream>>>(w2, W2, 256, 9, (size_t)256 * 2304);
    k_wconv<<<((256 * 256 ) + 255) / 256, blk, 0, stream>>>(w3, W3, 256, 1, (size_t)256 * 256);

    const int PK = NPOS * 16 / 256;   // pack grid
    const int EW = NPOS * 128 / 256;  // elementwise grid

    for (int g = 0; g < 4; ++g) {
        int zi = (g == 0);
        int yioff = zi ? 0 : 128 * (g - 1);
        // gates = conv3x3([yi, h])
        k_pack<<<PK, blk, 0, stream>>>(cat, 0,   qsd, 512, yioff, zi);
        k_pack<<<PK, blk, 0, stream>>>(cat, 128, hbuf, 128, 0, 0);
        k_gemm<0, true ><<<dim3(512, 4), blk, 0, stream>>>(Wl, cat, lstm_b, gates, nullptr, 512, 2304);
        k_lstm<<<EW, blk, 0, stream>>>(gates, cbuf, hbuf);
        // t = lrelu(conv1x1([h, yi])); t = lrelu(conv3x3(t)); cnd = conv1x1(t)
        k_pack<<<PK, blk, 0, stream>>>(cat, 0,   hbuf, 128, 0, 0);
        k_pack<<<PK, blk, 0, stream>>>(cat, 128, qsd, 512, yioff, zi);
        k_gemm<1, false><<<dim3(512, 2), blk, 0, stream>>>(W1, cat, b1, nullptr, t1, 256, 256);
        k_gemm<1, true ><<<dim3(512, 2), blk, 0, stream>>>(W2, t1,  b2, nullptr, t2, 256, 2304);
        k_gemm<0, false><<<dim3(512, 2), blk, 0, stream>>>(W3, t2,  b3, cnd, nullptr, 256, 256);
        k_lik<<<EW, blk, 0, stream>>>(cnd, qsd, lik, g);
    }
}
